// CausalSelfAttention_33054068310253
// MI455X (gfx1250) — matrix operand order repaired and checked by reading, not compiled, not run on hardware
//
#include <hip/hip_runtime.h>
#include <cstddef>
#include <cstdint>

// ---------------------------------------------------------------------------
// CDNA5 (gfx1250) causal self-attention, bf16 WMMA pipeline + TDM staging.
// Wave32; WMMA shape 16x16x32 bf16 -> f32 accumulate.
// ---------------------------------------------------------------------------

typedef __attribute__((ext_vector_type(16))) __bf16 bf16x16;
typedef __attribute__((ext_vector_type(8)))  __bf16 bf16x8;
typedef __attribute__((ext_vector_type(8)))  float  f32x8;
typedef __attribute__((ext_vector_type(4)))  unsigned int u32x4;
typedef __attribute__((ext_vector_type(8)))  int i32x8;
typedef __attribute__((ext_vector_type(4)))  int i32x4;

union Frag {            // 16 bf16 = 8 VGPRs
    bf16x16 v16;
    bf16x8  h[2];
};

__device__ __forceinline__ f32x8 zero8() {
    f32x8 z;
#pragma unroll
    for (int i = 0; i < 8; ++i) z[i] = 0.0f;
    return z;
}

__device__ __forceinline__ f32x8 wmma_bf16(const Frag& a, const Frag& b, f32x8 c) {
    return __builtin_amdgcn_wmma_f32_16x16x32_bf16(
        /*neg_a=*/false, a.v16, /*neg_b=*/false, b.v16,
        /*c_mod=*/(short)0, c, /*reuse_a=*/false, /*reuse_b=*/false);
}

// ---------------------------------------------------------------------------
// float -> bf16 conversion (RNE via hardware cvt)
// ---------------------------------------------------------------------------
__global__ __launch_bounds__(256)
void f32_to_bf16_kernel(const float* __restrict__ src, __bf16* __restrict__ dst, int n) {
    int i = blockIdx.x * blockDim.x + threadIdx.x;
    if (i < n) dst[i] = (__bf16)src[i];
}

// ---------------------------------------------------------------------------
// GEMM: C[M,N] = A[M,K] * B[N,K]^T   (A,B bf16 row-major, C f32 row-major)
// Block (8 waves) computes a 128x64 tile. The shared 64-row B panel (64 x K,
// K<=512 -> 64KB) is staged into LDS ONCE per block by the Tensor Data Mover:
// wave 0 issues one TENSOR_LOAD_TO_LDS descriptor (2-D tile, data_size=2B),
// waits on TENSORcnt, then the block barriers. Waves then run the K-loop with
// A fragments from global (L2-resident) and B fragments from LDS.
// All 4 B fragments are loaded before the 4 WMMAs so the LDS clause completes
// under one s_wait and the matrix ops issue back-to-back.
//
// Fragment layouts (A and B share one contiguous k order):
//   A (16x32): lane l -> row (l&15); 16 contiguous elements at k0+(l>>4)*16
//   B (32x16): lane l -> col (l&15); 16 contiguous elements at k0+(l>>4)*16
//   C (16x16): lane l -> col (l&15); VGPR i -> row (l>>4)*8 + i
// ---------------------------------------------------------------------------
__global__ __launch_bounds__(256)
void gemm_bf16_nt_tdm(const __bf16* __restrict__ A, const __bf16* __restrict__ B,
                      float* __restrict__ C, int M, int N, int K) {
    __shared__ __bf16 ldsB[64 * 512];            // 64 KB B panel (K <= 512)

    const int lane  = threadIdx.x & 31;
    const int wslot = threadIdx.x >> 5;          // 0..7
    const int tilesN = N >> 6;                   // N / 64
    const int bm = blockIdx.x / tilesN;          // 128-row block index
    const int nt = blockIdx.x % tilesN;          // 64-col tile index
    const int half = lane >> 4;
    const int ln   = lane & 15;

    // ---- TDM: stage B rows [nt*64, nt*64+64) x [0,K) into LDS ----
    if (wslot == 0) {
        const uint64_t gaddr = (uint64_t)(uintptr_t)(B + (size_t)nt * 64 * K);
        const uint32_t laddr = (uint32_t)(uintptr_t)(&ldsB[0]);  // LDS byte offset

        u32x4 g0;
        g0[0] = 1u;                                   // count=1, user descriptor
        g0[1] = laddr;                                // lds_addr  (bits 63:32)
        g0[2] = (uint32_t)gaddr;                      // global_addr[31:0]
        g0[3] = (uint32_t)(gaddr >> 32) | (2u << 30); // global_addr[56:32] | type=2

        const uint32_t td0 = (uint32_t)K;             // tensor_dim0 (elements)
        const uint32_t td1 = 64u;                     // tensor_dim1 (rows in tile)
        i32x8 g1;
        g1[0] = (int)(1u << 16);                      // wg_mask=0, data_size=1 (2B)
        g1[1] = (int)((td0 & 0xFFFFu) << 16);         // barrier_addr=0 | tensor_dim0 lo
        g1[2] = (int)((td0 >> 16) | ((td1 & 0xFFFFu) << 16)); // td0 hi | td1 lo
        g1[3] = (int)((td1 >> 16) | ((td0 & 0xFFFFu) << 16)); // td1 hi | tile_dim0=K
        g1[4] = 64;                                   // tile_dim1=64, tile_dim2=0
        g1[5] = (int)td0;                             // tensor_dim0_stride lo (=K)
        g1[6] = 0;                                    // stride0 hi | stride1 lo
        g1[7] = 0;                                    // stride1 hi

        i32x4 z4;
        z4[0] = z4[1] = z4[2] = z4[3] = 0;
#if defined(__clang_major__) && __clang_major__ >= 23
        i32x8 z8;
#pragma unroll
        for (int i = 0; i < 8; ++i) z8[i] = 0;
        __builtin_amdgcn_tensor_load_to_lds(g0, g1, z4, z4, z8, 0);
#else
        __builtin_amdgcn_tensor_load_to_lds(g0, g1, z4, z4, 0);
#endif
        __builtin_amdgcn_s_wait_tensorcnt(0);
    }
    __syncthreads();

    // ---- compute: this wave owns rows [bm*128 + wslot*16, +16) ----
    const int m0 = bm * 128 + wslot * 16;
    const __bf16* arow = A + (size_t)(m0 + ln) * K;

    f32x8 acc[4];
#pragma unroll
    for (int f = 0; f < 4; ++f) acc[f] = zero8();

    for (int kb = 0; kb < K; kb += 32) {
        Frag a;
        a.v16 = *(const bf16x16*)(arow + kb + half * 16);
        Frag b[4];
#pragma unroll
        for (int f = 0; f < 4; ++f)   // one LDS clause, one wait
            b[f].v16 = *(const bf16x16*)(&ldsB[(size_t)(f * 16 + ln) * K + kb + half * 16]);
#pragma unroll
        for (int f = 0; f < 4; ++f)   // 4 WMMAs back-to-back
            acc[f] = wmma_bf16(a, b[f], acc[f]);
    }

#pragma unroll
    for (int f = 0; f < 4; ++f) {
        float* ccol = C + (size_t)(m0 + half * 8) * N + nt * 64 + f * 16 + ln;
#pragma unroll
        for (int i = 0; i < 8; ++i) ccol[(size_t)i * N] = acc[f][i];
    }
}

// ---------------------------------------------------------------------------
// Fused epilogue on the QKV GEMM output (8192 x 1024 f32):
//   cols [0,512)    : Q (8 heads x 64)  -> RMSNorm, RoPE(16), *q_gain -> qb bf16 [B][8][T][64]
//   cols [512,768)  : K (4 heads x 64)  -> RMSNorm, RoPE(16)          -> kb bf16 [B][4][T][64]
//   cols [768,1024) : V (4 heads x 64)  -> convert, TRANSPOSED        -> vt bf16 [B][4][64][T]
// One thread per (token, head-slot); 16 slots per token.
// ---------------------------------------------------------------------------
__global__ __launch_bounds__(256)
void fuse_norm_rope_kernel(const float* __restrict__ qkv, const float* __restrict__ q_gain,
                           __bf16* __restrict__ qb, __bf16* __restrict__ kb,
                           __bf16* __restrict__ vt) {
    const int tid  = blockIdx.x * blockDim.x + threadIdx.x;   // 0 .. 8192*16-1
    const int slot = tid & 15;
    const int tok  = tid >> 4;                                // 0 .. 8191
    const int b = tok >> 11;                                  // / 2048
    const int t = tok & 2047;

    const float* src = qkv + (size_t)tok * 1024 + slot * 64;
    float v[64];
#pragma unroll
    for (int i = 0; i < 16; ++i) {
        float4 f = ((const float4*)src)[i];
        v[4 * i + 0] = f.x; v[4 * i + 1] = f.y;
        v[4 * i + 2] = f.z; v[4 * i + 3] = f.w;
    }

    if (slot < 12) {  // Q and K: RMSNorm over 64 dims + RoPE on first 16
        float ms = 0.0f;
#pragma unroll
        for (int i = 0; i < 64; ++i) ms += v[i] * v[i];
        float r = rsqrtf(ms * (1.0f / 64.0f) + 1.1920929e-7f);
#pragma unroll
        for (int i = 0; i < 64; ++i) v[i] *= r;

        const float ft = (float)t;
#pragma unroll
        for (int i = 0; i < 8; ++i) {
            // inv_freq = 10000^(-i/8) = exp(-ln(10000)/8 * i)
            float inv = __expf(-1.1512925465f * (float)i);
            float ang = ft * inv;
            float s, c;
            __sincosf(ang, &s, &c);
            float x1 = v[i], x2 = v[i + 8];
            v[i]     =  x1 * c + x2 * s;
            v[i + 8] = -x1 * s + x2 * c;
        }
    }

    if (slot < 8) {                       // Q head
        float g = q_gain[slot];
        __bf16* dst = qb + ((size_t)(b * 8 + slot) * 2048 + t) * 64;
#pragma unroll
        for (int i = 0; i < 64; ++i) dst[i] = (__bf16)(v[i] * g);
    } else if (slot < 12) {               // K head
        int kh = slot - 8;
        __bf16* dst = kb + ((size_t)(b * 4 + kh) * 2048 + t) * 64;
#pragma unroll
        for (int i = 0; i < 64; ++i) dst[i] = (__bf16)v[i];
    } else {                              // V head, store transposed (64 x T)
        int vh = slot - 12;
        __bf16* dst = vt + (size_t)(b * 4 + vh) * 64 * 2048;
#pragma unroll
        for (int i = 0; i < 64; ++i) dst[(size_t)i * 2048 + t] = (__bf16)v[i];
    }
}

// ---------------------------------------------------------------------------
// Flash attention: one wave per 16-query tile, online softmax, 32 keys/step.
//   S = Q(16x64) * K^T  : 4 v_wmma per step (all K frags loaded first)
//   V fragments are fetched right after the S-WMMAs so their global latency
//   overlaps the softmax VALU work.
//   P -> LDS (C-layout -> A-layout transpose) -> O += P(16x32) * V(32x64): 4 v_wmma
// GQA group = 2 (kv head = h >> 1). Output y in (b, t, h, d) bf16 for final GEMM.
// ---------------------------------------------------------------------------
#define FA_WAVES 8

__global__ __launch_bounds__(256)
void flash_attn_kernel(const __bf16* __restrict__ qb, const __bf16* __restrict__ kb,
                       const __bf16* __restrict__ vt, __bf16* __restrict__ yb) {
    __shared__ __bf16 ldsP[FA_WAVES][16 * 32];   // 1 KB per wave

    const int lane  = threadIdx.x & 31;
    const int wslot = threadIdx.x >> 5;
    const int wid   = blockIdx.x * FA_WAVES + wslot;   // 0 .. 4095
    const int qt  = wid & 127;      // 128 q-tiles per (b,h)
    const int bh  = wid >> 7;
    const int b   = bh >> 3;
    const int h   = bh & 7;
    const int kvh = h >> 1;
    const int half = lane >> 4;
    const int ln   = lane & 15;
    const int q0   = qt * 16;

    const __bf16* Qp = qb + (size_t)(b * 8 + h)   * 2048 * 64;
    const __bf16* Kp = kb + (size_t)(b * 4 + kvh) * 2048 * 64;
    const __bf16* Vt = vt + (size_t)(b * 4 + kvh) * 64 * 2048;

    // Q fragments for k = 0..31 and 32..63
    Frag qa[2];
    {
        const __bf16* qrow = Qp + (size_t)(q0 + ln) * 64;
#pragma unroll
        for (int c = 0; c < 2; ++c) {
            qa[c].v16 = *(const bf16x16*)(qrow + c * 32 + half * 16);
        }
    }

    float mrow[8], lrow[8];
    f32x8 o[4];
#pragma unroll
    for (int i = 0; i < 8; ++i) { mrow[i] = -3.0e38f; lrow[i] = 0.0f; }
#pragma unroll
    for (int f = 0; f < 4; ++f) o[f] = zero8();

    __bf16* myP = &ldsP[wslot][0];

    for (int j32 = 0; j32 < q0 + 16; j32 += 32) {
        if (j32 + 32 < 2048)  // speculative prefetch of next K tile -> global_prefetch_b8
            __builtin_prefetch(Kp + (size_t)(j32 + 32 + ln) * 64, 0, 1);

        // ---- load all 4 K fragments (one global clause) ----
        Frag kf[2][2];
#pragma unroll
        for (int t2 = 0; t2 < 2; ++t2) {
            const __bf16* krow = Kp + (size_t)(j32 + t2 * 16 + ln) * 64;
#pragma unroll
            for (int c = 0; c < 2; ++c)
                kf[t2][c].v16 = *(const bf16x16*)(krow + c * 32 + half * 16);
        }

        // ---- S = Q * K^T (two 16x16 score frags covering 32 keys) ----
        f32x8 s[2];
#pragma unroll
        for (int t2 = 0; t2 < 2; ++t2) {
            s[t2] = zero8();
            s[t2] = wmma_bf16(qa[0], kf[t2][0], s[t2]);
            s[t2] = wmma_bf16(qa[1], kf[t2][1], s[t2]);
        }

        // ---- V fragments now: latency hides under the softmax below ----
        Frag vf[4];
#pragma unroll
        for (int f = 0; f < 4; ++f)
            vf[f].v16 = *(const bf16x16*)(Vt + (size_t)(f * 16 + ln) * 2048 + j32 + half * 16);

        // ---- scale + causal mask + online softmax ----
#pragma unroll
        for (int i = 0; i < 8; ++i) {
            const int qm = q0 + half * 8 + i;
#pragma unroll
            for (int t2 = 0; t2 < 2; ++t2) {
                const int key = j32 + t2 * 16 + ln;
                float val = s[t2][i] * 0.125f;               // 1/sqrt(64)
                s[t2][i] = (key <= qm) ? val : -3.0e38f;
            }
            float v = fmaxf(s[0][i], s[1][i]);
#pragma unroll
            for (int off = 1; off < 16; off <<= 1)
                v = fmaxf(v, __shfl_xor(v, off, 32));        // row-max within 16-lane half
            float mnew  = fmaxf(mrow[i], v);
            float alpha = __expf(mrow[i] - mnew);
            mrow[i] = mnew;
            lrow[i] *= alpha;
#pragma unroll
            for (int f = 0; f < 4; ++f) o[f][i] *= alpha;
            float rsum = 0.0f;
#pragma unroll
            for (int t2 = 0; t2 < 2; ++t2) {
                float p = __expf(s[t2][i] - mnew);
                s[t2][i] = p;
                rsum += p;
            }
#pragma unroll
            for (int off = 1; off < 16; off <<= 1)
                rsum += __shfl_xor(rsum, off, 32);
            lrow[i] += rsum;
        }

        // ---- P: C-layout -> LDS -> A-layout ----
#pragma unroll
        for (int i = 0; i < 8; ++i) {
            const int m = half * 8 + i;
            myP[m * 32 + ln]      = (__bf16)s[0][i];
            myP[m * 32 + 16 + ln] = (__bf16)s[1][i];
        }
        asm volatile("s_wait_dscnt 0" ::: "memory");  // wave-local LDS RAW fence

        Frag pa;
        {
            const __bf16* prow = myP + ln * 32;
            pa.h[0] = *(const bf16x8*)(prow + half * 16);
            pa.h[1] = *(const bf16x8*)(prow + half * 16 + 8);
        }

        // ---- O += P * V : 4 WMMAs back-to-back ----
#pragma unroll
        for (int f = 0; f < 4; ++f)
            o[f] = wmma_bf16(pa, vf[f], o[f]);
    }

    // ---- normalize + store y as (b, t, h, d) bf16 ----
#pragma unroll
    for (int i = 0; i < 8; ++i) {
        const float inv = 1.0f / lrow[i];
        const int q = q0 + half * 8 + i;
        __bf16* yrow = yb + ((size_t)(b * 2048 + q) * 8 + h) * 64;
#pragma unroll
        for (int f = 0; f < 4; ++f)
            yrow[f * 16 + ln] = (__bf16)(o[f][i] * inv);
    }
}

// ---------------------------------------------------------------------------
// Host-side orchestration
// ---------------------------------------------------------------------------
extern "C" void kernel_launch(void* const* d_in, const int* in_sizes, int n_in,
                              void* d_out, int out_size, void* d_ws, size_t ws_size,
                              hipStream_t stream) {
    (void)in_sizes; (void)n_in; (void)out_size; (void)ws_size;

    const float* x      = (const float*)d_in[0];   // (4, 2048, 512)
    const float* w_q    = (const float*)d_in[1];   // (512, 512)
    const float* w_k    = (const float*)d_in[2];   // (256, 512)
    const float* w_v    = (const float*)d_in[3];   // (256, 512)
    const float* w_proj = (const float*)d_in[4];   // (512, 512)
    const float* q_gain = (const float*)d_in[5];   // (8,)
    float* out = (float*)d_out;                    // (4, 2048, 512) f32

    const int BT = 4 * 2048;      // 8192 tokens
    const int D  = 512;
    const int NQKV = 1024;        // 512 Q + 256 K + 256 V output features

    // workspace carve-up
    char* ws = (char*)d_ws;
    __bf16* xb   = (__bf16*)ws;                       // 8192 x 512
    __bf16* wb   = xb + (size_t)BT * D;               // 1024 x 512 (Q,K,V rows packed)
    __bf16* wpb  = wb + (size_t)NQKV * D;             // 512 x 512
    float*  qkvf = (float*)(wpb + (size_t)D * D);     // 8192 x 1024 f32
    __bf16* qb   = (__bf16*)(qkvf + (size_t)BT * NQKV);     // [4][8][2048][64]
    __bf16* kbuf = qb + (size_t)4 * 8 * 2048 * 64;          // [4][4][2048][64]
    __bf16* vt   = kbuf + (size_t)4 * 4 * 2048 * 64;        // [4][4][64][2048]
    __bf16* yb   = vt + (size_t)4 * 4 * 64 * 2048;          // 8192 x 512

    // 1) convert inputs to bf16 (weights packed as rows of one (1024 x 512) B matrix)
    f32_to_bf16_kernel<<<(BT * D + 255) / 256, 256, 0, stream>>>(x, xb, BT * D);
    f32_to_bf16_kernel<<<(512 * 512 + 255) / 256, 256, 0, stream>>>(w_q, wb, 512 * 512);
    f32_to_bf16_kernel<<<(256 * 512 + 255) / 256, 256, 0, stream>>>(w_k, wb + 512 * 512, 256 * 512);
    f32_to_bf16_kernel<<<(256 * 512 + 255) / 256, 256, 0, stream>>>(w_v, wb + 768 * 512, 256 * 512);
    f32_to_bf16_kernel<<<(512 * 512 + 255) / 256, 256, 0, stream>>>(w_proj, wpb, 512 * 512);

    // 2) fused QKV projection: (8192 x 512) * (1024 x 512)^T -> 8192 x 1024 f32
    {
        int blocks = (BT / 128) * (NQKV / 64);  // 64 * 16 = 1024 blocks of 8 waves
        gemm_bf16_nt_tdm<<<blocks, 256, 0, stream>>>(xb, wb, qkvf, BT, NQKV, D);
    }

    // 3) RMSNorm + RoPE + gain + layout (V transposed)
    fuse_norm_rope_kernel<<<(BT * 16) / 256, 256, 0, stream>>>(qkvf, q_gain, qb, kbuf, vt);

    // 4) flash attention: 4 batches x 8 heads x 128 q-tiles = 4096 waves
    flash_attn_kernel<<<4096 / FA_WAVES, 256, 0, stream>>>(qb, kbuf, vt, yb);

    // 5) output projection: (8192 x 512) * (512 x 512)^T -> d_out f32
    {
        int blocks = (BT / 128) * (D / 64);     // 64 * 8 = 512 blocks of 8 waves
        gemm_bf16_nt_tdm<<<blocks, 256, 0, stream>>>(yb, wpb, out, BT, D, D);
    }
}
